// PermutohedralLayer_51419348468015
// MI455X (gfx1250) — matrix operand order repaired and checked by reading, not compiled, not run on hardware
//
#include <hip/hip_runtime.h>
#include <hip/hip_bf16.h>

// CDNA5 / gfx1250 permutohedral layer: splat -> WMMA blur -> slice.
// Blur uses v_wmma_f32_16x16x32_bf16 (16x16 f32 acc, K=32 = C_IN per tap),
// LDS-staged neighbor indices, branch-free gathers.

typedef __attribute__((ext_vector_type(16))) __bf16 v16bf;
typedef __attribute__((ext_vector_type(8)))  float  v8f;

#define C_IN  32
#define C_OUT 32
#define D1    6      // barycentric dims (d+1)
#define FMAX  64     // max taps supported by the LDS fast path (F = 2^D1 - 1 = 63)

// ---- helpers -------------------------------------------------------------

__device__ __forceinline__ unsigned short f2bf(float f) {
  unsigned int u = __builtin_bit_cast(unsigned int, f);
  u += 0x7FFFu + ((u >> 16) & 1u);   // round-to-nearest-even
  return (unsigned short)(u >> 16);
}

// ---- kernel 1: zero splat buffer ----------------------------------------

__global__ void k_zero(float* __restrict__ p, long n) {
  long i = (long)blockIdx.x * blockDim.x + threadIdx.x;
  if (i < n) p[i] = 0.0f;
}

// ---- kernel 2: splat (scatter-add into lattice vertices) ----------------
// V layout: V[m][c], row-major per vertex (32 contiguous channels).

__global__ void k_splat(const float* __restrict__ data,
                        const float* __restrict__ bary,
                        const int*   __restrict__ sidx,
                        float* __restrict__ V, int N) {
  long i = (long)blockIdx.x * blockDim.x + threadIdx.x;
  if (i >= (long)N * C_IN) return;
  int n = (int)(i % N);
  int c = (int)(i / N);
  float x = data[i];                       // data[c*N + n]
  #pragma unroll
  for (int k = 0; k < D1; ++k) {
    int   m = sidx[(long)n * D1 + k];
    float w = bary[(long)n * D1 + k];
    atomicAdd(&V[(long)m * C_IN + c], x * w);
  }
}

// ---- kernel 3: convert V (f32) -> Vh (bf16) -----------------------------

__global__ void k_cvt(const float* __restrict__ V,
                      unsigned short* __restrict__ Vh, long n) {
  long i = (long)blockIdx.x * blockDim.x + threadIdx.x;
  if (i < n) Vh[i] = f2bf(V[i]);
}

// ---- kernel 4: pack weights into WMMA A-operand lane layout -------------
// Wpack[f][oblk][lane][v] (dwords).  16-bit A 16x32, contiguous k order used here (B uses the same):
//   lanes 0-15  : M=lane,    K={2v,2v+1}
//   lanes 16-31 : M=lane-16, K={16+2v,17+2v}

__global__ void k_packw(const float* __restrict__ w,   // weights[o][c][0][f]
                        unsigned int* __restrict__ wp, int F) {
  int t = blockIdx.x * blockDim.x + threadIdx.x;
  int total = F * 2 * 32;
  if (t >= total) return;
  int lane = t & 31;
  int oblk = (t >> 5) & 1;
  int f    = t >> 6;
  int o     = oblk * 16 + (lane & 15);
  int khalf = (lane >> 4) ? 16 : 0;
  unsigned int* dst = wp + (((long)f * 2 + oblk) * 32 + lane) * 8;
  #pragma unroll
  for (int v = 0; v < 8; ++v) {
    int kb = 2 * v + khalf;
    float w0 = w[((long)o * C_IN + kb    ) * F + f];
    float w1 = w[((long)o * C_IN + kb + 1) * F + f];
    dst[v] = (unsigned int)f2bf(w0) | ((unsigned int)f2bf(w1) << 16);
  }
}

// ---- kernel 5: blur — the WMMA core -------------------------------------
// One wave32 per (16-vertex tile, 16-output-channel block).
// Per tap f: A = packed W_f (16x32 bf16), B = gathered vertex rows (32x16),
// acc(16x16 f32) accumulated over all F taps with v_wmma_f32_16x16x32_bf16.
// Neighbor indices for the tile are staged in LDS (one coalesced copy of the
// contiguous 16xF row-major block), so the tap loop is branch-free:
//   1 ds_load_b32 (index) + 2 b128 (vertex gather) + 2 b128 (weights) + wmma.

__global__ __launch_bounds__(256) void k_blur(
    const unsigned short* __restrict__ Vh,   // (M+1) x 32 bf16, row M == 0
    const int*   __restrict__ nb,            // M x F neighbor indices
    const unsigned int* __restrict__ wp,     // packed weights
    float* __restrict__ outl,                // M x 32 f32, per-vertex rows
    int M, int F) {
  __shared__ int ldsIdx[8][16 * FMAX];       // 32 KB, one slot per wave

  int wslot = threadIdx.x >> 5;
  int lane  = threadIdx.x & 31;
  int wave  = blockIdx.x * 8 + wslot;
  int numTiles = (M + 15) >> 4;
  int oblk = wave & 1;
  int tile = wave >> 1;
  if (tile >= numTiles) return;

  int  col   = lane & 15;                 // B-matrix column / C-matrix column
  int  m     = tile * 16 + col;
  int  kbase = (lane >> 4) << 4;          // lanes 16-31 hold K=16..31
  bool valid = (m < M);
  int  mi    = valid ? m : (M - 1);       // clamped, always-safe row

  const unsigned int* wbase = wp + (long)oblk * 32 * 8 + (long)lane * 8;
  const long wstride = 2L * 32 * 8;       // dwords per tap
  v8f acc = {};

  if (F <= FMAX) {
    // ---- stage the tile's contiguous 16xF index block into LDS ----
    {
      long base  = (long)tile * 16 * F;
      long limit = (long)M * F;
      int  count = 16 * F;
      for (int j = lane; j < count; j += 32) {
        long g = base + j;
        ldsIdx[wslot][j] = (g < limit) ? nb[g] : M;
      }
    }
    const int* idxRow = &ldsIdx[wslot][col * F];
    for (int f = 0; f < F; ++f) {
      int idx = idxRow[f];                          // ds_load_b32
      idx = valid ? idx : M;                        // cndmask, no branch
      v16bf bmat = *(const v16bf*)(Vh + (long)idx * C_IN + kbase);
      const unsigned int* wl = wbase + (long)f * wstride;
      v16bf amat = *(const v16bf*)wl;
      __builtin_prefetch(wl + wstride, 0, 3);       // near-cache prefetch
      acc = __builtin_amdgcn_wmma_f32_16x16x32_bf16(
          false, amat, false, bmat, (short)0, acc, false, false);
    }
  } else {
    // ---- fallback: direct (clamped) global index loads ----
    const int* nrow = nb + (long)mi * F;
    for (int f = 0; f < F; ++f) {
      int idx = nrow[f];
      idx = valid ? idx : M;
      v16bf bmat = *(const v16bf*)(Vh + (long)idx * C_IN + kbase);
      v16bf amat = *(const v16bf*)(wbase + (long)f * wstride);
      acc = __builtin_amdgcn_wmma_f32_16x16x32_bf16(
          false, amat, false, bmat, (short)0, acc, false, false);
    }
  }

  if (valid) {
    // C/D layout: VGPR r, lane l -> row (r + 8*(l>=16)), col (l&15).
    // Lane's 8 accumulators are contiguous output channels -> 2x float4.
    float* dst = outl + (long)m * C_OUT + oblk * 16 + ((lane >> 4) << 3);
    float4 lo = make_float4(acc[0], acc[1], acc[2], acc[3]);
    float4 hi = make_float4(acc[4], acc[5], acc[6], acc[7]);
    *(float4*)(dst)     = lo;
    *(float4*)(dst + 4) = hi;
  }
}

// ---- kernel 6: slice + bias ---------------------------------------------
// o-fast thread mapping: each wave reads the six gathered vertex rows as
// coalesced 128B transactions (gather volume >> write volume).

__global__ void k_slice(const float* __restrict__ outl,
                        const int*   __restrict__ sidx,
                        const float* __restrict__ bary,
                        const float* __restrict__ bias,
                        const float* __restrict__ bmul,
                        float* __restrict__ y, int N, float alpha) {
  long i = (long)blockIdx.x * blockDim.x + threadIdx.x;
  if (i >= (long)N * C_OUT) return;
  int o = (int)(i & (C_OUT - 1));
  int n = (int)(i >> 5);
  float s = 0.0f;
  #pragma unroll
  for (int k = 0; k < D1; ++k) {
    int m = sidx[(long)n * D1 + k];
    s += outl[(long)m * C_OUT + o] * bary[(long)n * D1 + k];
  }
  y[(long)o * N + n] = alpha * s + bias[o] * bmul[n];
}

// ---- launch --------------------------------------------------------------

extern "C" void kernel_launch(void* const* d_in, const int* in_sizes, int n_in,
                              void* d_out, int out_size, void* d_ws, size_t ws_size,
                              hipStream_t stream) {
  const float* data    = (const float*)d_in[0];
  const float* weights = (const float*)d_in[1];
  const float* bias    = (const float*)d_in[2];
  const float* bmul    = (const float*)d_in[3];
  const float* bary    = (const float*)d_in[4];
  const int*   sidx    = (const int*)d_in[5];
  const int*   nbrs    = (const int*)d_in[6];
  float*       out     = (float*)d_out;

  const int F = in_sizes[1] / (C_OUT * C_IN);   // taps (63)
  const int N = in_sizes[0] / C_IN;             // pixels (B==1)
  const int M = in_sizes[6] / F;                // lattice vertices

  // carve workspace (256B-aligned regions)
  char* ws = (char*)d_ws;
  size_t off = 0;
  auto carve = [&](size_t bytes) -> char* {
    char* p = ws + off;
    off = (off + bytes + 255) & ~(size_t)255;
    return p;
  };
  float*          V    = (float*)         carve((size_t)(M + 1) * C_IN * sizeof(float));
  unsigned short* Vh   = (unsigned short*)carve((size_t)(M + 1) * C_IN * sizeof(unsigned short));
  float*          outl = (float*)         carve((size_t)M * C_OUT * sizeof(float));
  unsigned int*   wp   = (unsigned int*)  carve((size_t)F * 2 * 32 * 8 * sizeof(unsigned int));

  const int T = 256;
  const float alpha = 1.0f / (1.0f + exp2f(-(float)(D1 - 1)));   // 32/33

  // 1) zero V (including the zero row M used for missing neighbors)
  {
    long n = (long)(M + 1) * C_IN;
    k_zero<<<dim3((unsigned)((n + T - 1) / T)), dim3(T), 0, stream>>>(V, n);
  }
  // 2) splat
  {
    long n = (long)N * C_IN;
    k_splat<<<dim3((unsigned)((n + T - 1) / T)), dim3(T), 0, stream>>>(data, bary, sidx, V, N);
  }
  // 3) bf16 convert
  {
    long n = (long)(M + 1) * C_IN;
    k_cvt<<<dim3((unsigned)((n + T - 1) / T)), dim3(T), 0, stream>>>(V, Vh, n);
  }
  // 4) pack weights into WMMA A-operand layout
  {
    int n = F * 2 * 32;
    k_packw<<<dim3((n + T - 1) / T), dim3(T), 0, stream>>>(weights, wp, F);
  }
  // 5) blur: one wave per (tile, oblk)
  {
    int numTiles = (M + 15) >> 4;
    long threads = (long)numTiles * 2 * 32;
    k_blur<<<dim3((unsigned)((threads + T - 1) / T)), dim3(T), 0, stream>>>(
        Vh, nbrs, wp, outl, M, F);
  }
  // 6) slice + bias
  {
    long n = (long)N * C_OUT;
    k_slice<<<dim3((unsigned)((n + T - 1) / T)), dim3(T), 0, stream>>>(
        outl, sidx, bary, bias, bmul, out, N, alpha);
  }
}
